// PatternsOfThinking_6305011990652
// MI455X (gfx1250) — compile-verified, matrix operand order checked by reading, not run on hardware
//
#include <hip/hip_runtime.h>
#include <stdint.h>

// ---------------------------------------------------------------------------
// PatternsOfThinking transformer block for MI455X (gfx1250, wave32, WMMA).
// GEMMs run as bf16 x bf16 -> f32 via V_WMMA_F32_16X16X32_BF16.
// Reductions (argmax / softmax / layernorm / pot mix) stay fp32.
// ---------------------------------------------------------------------------

#define DEVFN __device__ __forceinline__

typedef __attribute__((ext_vector_type(16))) __bf16 v16bf;
typedef __attribute__((ext_vector_type(8)))  float  v8f;

constexpr int Bx  = 2;
constexpr int Sx  = 2048;
constexpr int Ex  = 512;
constexpr int Hx  = 8;
constexpr int HDx = 64;
constexpr int FEx = 4 * Ex;          // 2048
constexpr int MSx = Bx * Sx;         // 4096 rows

DEVFN uint16_t f2bf(float f) {
  uint32_t u = __float_as_uint(f);
  u += 0x7FFFu + ((u >> 16) & 1u);   // round-to-nearest-even
  return (uint16_t)(u >> 16);
}

DEVFN float gelu_exact(float x) {
  return 0.5f * x * (1.0f + erff(x * 0.70710678118654752f));
}

// ------------------------- prep kernels ------------------------------------

__global__ __launch_bounds__(256) void cvt_bf16_kernel(
    const float* __restrict__ src, uint16_t* __restrict__ dst, int n) {
  int i = blockIdx.x * 256 + threadIdx.x;
  if (i < n) dst[i] = f2bf(src[i]);
}

// W [K x N] row-major f32  ->  Wt [N x K] bf16 (column-major for WMMA B)
__global__ __launch_bounds__(256) void transpose_bf16_kernel(
    const float* __restrict__ W, uint16_t* __restrict__ Wt, int K, int N) {
  int i = blockIdx.x * 256 + threadIdx.x;
  if (i >= K * N) return;
  int k = i / N, n = i - k * N;
  Wt[(size_t)n * K + k] = f2bf(W[i]);
}

// ------------------------- WMMA GEMM ---------------------------------------
// C[MxN] = A[MxK] * B[KxN] *scale (+bias), A row-major bf16 (lda), B supplied
// column-major bf16 (Bt[n][k], ldb). One 16x16 tile per wave, 8 waves/WG.
// Per-lane fragment layout per CDNA5 ISA 7.12.2 (16-bit A/B 16x32):
//   lane = 16*h + lr ; vgpr r holds K pair at koff = 2r + 8h + (r>=4 ? 8 : 0)

constexpr int OUT_F32       = 0;  // f32, [row*ldc+col] (+ z*sCz)
constexpr int OUT_BF16_GELU = 1;  // gelu -> bf16
constexpr int OUT_QK        = 2;  // bf16 -> [B,H,S,HD]
constexpr int OUT_VT        = 3;  // bf16 -> [B,H,HD,S] (transposed V)
constexpr int OUT_MERGE     = 4;  // bf16 -> [B,S,E] head-merge (z = b*H+h)

template <int OMODE>
__global__ __launch_bounds__(256) void gemm_wmma_bf16(
    const uint16_t* __restrict__ A, long long sAz, int lda,
    const uint16_t* __restrict__ Bt, long long sBz, int ldb,
    const float* __restrict__ bias,
    void* __restrict__ Cout, long long sCz, int ldc,
    int Mtiles, int Ntiles, int Kdim, float scale) {
  const int z = blockIdx.z;
  A  += (size_t)z * sAz;
  Bt += (size_t)z * sBz;

  const int wave = threadIdx.x >> 5;
  const int lane = threadIdx.x & 31;
  const int tile = blockIdx.x * 8 + wave;
  if (tile >= Mtiles * Ntiles) return;
  const int tm = tile / Ntiles;
  const int tn = tile - tm * Ntiles;

  const int h  = lane >> 4;    // half-wave
  const int lr = lane & 15;    // A row / B col within tile

  const uint16_t* Arow = A  + (size_t)(tm * 16 + lr) * lda;
  const uint16_t* Brow = Bt + (size_t)(tn * 16 + lr) * ldb;

  union Frag { v16bf v; uint32_t u[8]; };
  v8f acc = {0.f, 0.f, 0.f, 0.f, 0.f, 0.f, 0.f, 0.f};

  for (int k0 = 0; k0 < Kdim; k0 += 32) {
    if (k0 + 32 < Kdim) {                       // hint next K-slab into cache
      __builtin_prefetch(Arow + k0 + 32, 0, 1);
      __builtin_prefetch(Brow + k0 + 32, 0, 1);
    }
    Frag a, b;
#pragma unroll
    for (int r = 0; r < 8; ++r) {
      const int koff = 2 * r + 8 * h + ((r >= 4) ? 8 : 0);
      a.u[r] = *reinterpret_cast<const uint32_t*>(Arow + k0 + koff);
      b.u[r] = *reinterpret_cast<const uint32_t*>(Brow + k0 + koff);
    }
    acc = __builtin_amdgcn_wmma_f32_16x16x32_bf16(
        false, a.v, false, b.v, (short)0, acc, false, false);
  }

  const int col = tn * 16 + lr;
  const float bv = bias ? bias[col] : 0.f;
#pragma unroll
  for (int i = 0; i < 8; ++i) {
    const int row = tm * 16 + i + 8 * h;        // C/D layout: M = i + 8*half
    float v = acc[i] * scale + bv;
    if constexpr (OMODE == OUT_F32) {
      ((float*)Cout)[(size_t)z * sCz + (size_t)row * ldc + col] = v;
    } else if constexpr (OMODE == OUT_BF16_GELU) {
      ((uint16_t*)Cout)[(size_t)row * ldc + col] = f2bf(gelu_exact(v));
    } else if constexpr (OMODE == OUT_QK) {
      const int b = row / Sx, s = row % Sx, hh = col / HDx, d = col % HDx;
      ((uint16_t*)Cout)[(((size_t)(b * Hx + hh)) * Sx + s) * HDx + d] = f2bf(v);
    } else if constexpr (OMODE == OUT_VT) {
      const int b = row / Sx, s = row % Sx, hh = col / HDx, d = col % HDx;
      ((uint16_t*)Cout)[(((size_t)(b * Hx + hh)) * HDx + d) * Sx + s] = f2bf(v);
    } else {  // OUT_MERGE: z = b*H + h, row = q, col = d
      const int b = z / Hx, hh = z - b * Hx;
      ((uint16_t*)Cout)[((size_t)b * Sx + row) * Ex + hh * HDx + col] = f2bf(v);
    }
  }
}

// ------------------------- pot block ---------------------------------------

// per-row max + argmax (first occurrence on ties, like jnp.argmax)
__global__ __launch_bounds__(256) void rowmax_argmax_kernel(
    const float* __restrict__ scores, float* __restrict__ mv, int* __restrict__ idx) {
  const int row = blockIdx.x;
  const float* srow = scores + (size_t)row * Sx;
  const int t = threadIdx.x;
  float best = -3.4e38f; int bi = 0;
  for (int c = t; c < Sx; c += 256) {
    float v = srow[c];
    if (v > best) { best = v; bi = c; }
  }
  __shared__ float sv[256]; __shared__ int si[256];
  sv[t] = best; si[t] = bi;
  __syncthreads();
  for (int off = 128; off > 0; off >>= 1) {
    if (t < off) {
      float ov = sv[t + off]; int oi = si[t + off];
      if (ov > sv[t] || (ov == sv[t] && oi < si[t])) { sv[t] = ov; si[t] = oi; }
    }
    __syncthreads();
  }
  if (t == 0) { mv[row] = sv[0]; idx[row] = si[0]; }
}

// tv[m][j] = gelu( sum_s mv[m][s] * Wp[s][j] + bp[j] ),  m = b*H+h (16 rows)
__global__ __launch_bounds__(256) void pot_transform_kernel(
    const float* __restrict__ mv, const float* __restrict__ Wp,
    const float* __restrict__ bp, float* __restrict__ tv) {
  const int j = blockIdx.x * 256 + threadIdx.x;
  __shared__ float smv[16][257];
  float acc[16];
#pragma unroll
  for (int m = 0; m < 16; ++m) acc[m] = 0.f;
  for (int s0 = 0; s0 < Sx; s0 += 256) {
#pragma unroll
    for (int m = 0; m < 16; ++m) smv[m][threadIdx.x] = mv[m * Sx + s0 + threadIdx.x];
    __syncthreads();
    for (int ss = 0; ss < 256; ++ss) {
      const float w = Wp[(size_t)(s0 + ss) * Sx + j];  // coalesced over j
#pragma unroll
      for (int m = 0; m < 16; ++m) acc[m] += smv[m][ss] * w;
    }
    __syncthreads();
  }
  const float b = bp[j];
#pragma unroll
  for (int m = 0; m < 16; ++m) tv[m * Sx + j] = gelu_exact(acc[m] + b);
}

// substitute transformed value at argmax, softmax, emit bf16 attn row.
// NOTE: attn aliases the score buffer (bf16 row packed at the start of each
// 8KB f32 row) -- safe: the full row is read into registers, barriers sit
// between load and store phases, and no __restrict__ on aliasing pointers.
__global__ __launch_bounds__(256) void softmax_sub_kernel(
    const float* scores, uint16_t* attn,
    const int* __restrict__ idx, const float* __restrict__ tv) {
  const int row = blockIdx.x;              // (b*H+h)*S + q
  const int bh = row / Sx, q = row - bh * Sx;
  const float tval = tv[bh * Sx + q];
  const int ii = idx[row];
  const float* srow = scores + (size_t)row * Sx;
  uint16_t* arow = attn + (size_t)row * (2 * Sx);   // lda = 4096 bf16
  const int t = threadIdx.x;

  float v[8]; float lmax = -3.4e38f;
#pragma unroll
  for (int i = 0; i < 8; ++i) {
    const int c = t + i * 256;
    float s = srow[c];
    if (c == ii) s = tval;
    v[i] = s;
    lmax = fmaxf(lmax, s);
  }
  __shared__ float red[256];
  red[t] = lmax; __syncthreads();
  for (int off = 128; off > 0; off >>= 1) {
    if (t < off) red[t] = fmaxf(red[t], red[t + off]);
    __syncthreads();
  }
  const float rmax = red[0];
  __syncthreads();
  float lsum = 0.f;
#pragma unroll
  for (int i = 0; i < 8; ++i) { v[i] = __expf(v[i] - rmax); lsum += v[i]; }
  red[t] = lsum; __syncthreads();
  for (int off = 128; off > 0; off >>= 1) {
    if (t < off) red[t] += red[t + off];
    __syncthreads();
  }
  const float inv = 1.0f / red[0];
#pragma unroll
  for (int i = 0; i < 8; ++i) arow[t + i * 256] = f2bf(v[i] * inv);
}

// ------------------------- residual + layernorm ----------------------------

__global__ __launch_bounds__(256) void add_ln_kernel(
    const float* __restrict__ ffn, const float* __restrict__ x,
    const float* __restrict__ gamma, const float* __restrict__ beta,
    float* __restrict__ out) {
  const int row = blockIdx.x;
  const int t = threadIdx.x;
  const size_t base = (size_t)row * Ex;
  float a0 = ffn[base + t]       + x[base + t];
  float a1 = ffn[base + t + 256] + x[base + t + 256];
  __shared__ float red[256];
  red[t] = a0 + a1; __syncthreads();
  for (int off = 128; off > 0; off >>= 1) {
    if (t < off) red[t] += red[t + off];
    __syncthreads();
  }
  const float mu = red[0] * (1.0f / Ex);
  __syncthreads();
  const float d0 = a0 - mu, d1 = a1 - mu;
  red[t] = d0 * d0 + d1 * d1; __syncthreads();
  for (int off = 128; off > 0; off >>= 1) {
    if (t < off) red[t] += red[t + off];
    __syncthreads();
  }
  const float w = rsqrtf(red[0] * (1.0f / Ex) + 1e-5f);
  out[base + t]       = d0 * w * gamma[t]       + beta[t];
  out[base + t + 256] = d1 * w * gamma[t + 256] + beta[t + 256];
}

// ------------------------- launch ------------------------------------------

extern "C" void kernel_launch(void* const* d_in, const int* in_sizes, int n_in,
                              void* d_out, int out_size, void* d_ws, size_t ws_size,
                              hipStream_t stream) {
  (void)in_sizes; (void)n_in; (void)out_size; (void)ws_size;
  const float* x   = (const float*)d_in[0];
  const float* W1  = (const float*)d_in[1];
  const float* b1  = (const float*)d_in[2];
  const float* W2  = (const float*)d_in[3];
  const float* b2  = (const float*)d_in[4];
  const float* W3  = (const float*)d_in[5];
  const float* b3  = (const float*)d_in[6];
  const float* Wp  = (const float*)d_in[7];
  const float* bp  = (const float*)d_in[8];
  const float* Wf1 = (const float*)d_in[9];
  const float* bf1 = (const float*)d_in[10];
  const float* Wf2 = (const float*)d_in[11];
  const float* bf2 = (const float*)d_in[12];
  const float* gamma = (const float*)d_in[13];
  const float* beta  = (const float*)d_in[14];

  char* ws = (char*)d_ws;
  size_t off = 0;
  auto alloc = [&](size_t bytes) -> void* {
    void* p = ws + off;
    off += (bytes + 255) & ~(size_t)255;
    return p;
  };

  uint16_t* xbf   = (uint16_t*)alloc((size_t)MSx * Ex * 2);          //  4 MB
  uint16_t* W1t   = (uint16_t*)alloc((size_t)Ex * Ex * 2);
  uint16_t* W2t   = (uint16_t*)alloc((size_t)Ex * Ex * 2);
  uint16_t* W3t   = (uint16_t*)alloc((size_t)Ex * Ex * 2);
  uint16_t* Wf1t  = (uint16_t*)alloc((size_t)FEx * Ex * 2);          //  2 MB
  uint16_t* Wf2t  = (uint16_t*)alloc((size_t)Ex * FEx * 2);          //  2 MB
  uint16_t* Qb    = (uint16_t*)alloc((size_t)Bx * Hx * Sx * HDx * 2);
  uint16_t* Kb    = (uint16_t*)alloc((size_t)Bx * Hx * Sx * HDx * 2);
  uint16_t* Vt    = (uint16_t*)alloc((size_t)Bx * Hx * HDx * Sx * 2);
  float*    scores= (float*)   alloc((size_t)Bx * Hx * Sx * Sx * 4); // 256 MB
  float*    mv    = (float*)   alloc((size_t)Bx * Hx * Sx * 4);
  int*      idx   = (int*)     alloc((size_t)Bx * Hx * Sx * 4);
  float*    tv    = (float*)   alloc((size_t)Bx * Hx * Sx * 4);
  uint16_t* attO  = (uint16_t*)alloc((size_t)MSx * Ex * 2);          //  4 MB
  uint16_t* h1    = (uint16_t*)alloc((size_t)MSx * FEx * 2);         // 16 MB
  float*    ffn   = (float*)   alloc((size_t)MSx * Ex * 4);          //  8 MB
  uint16_t* attn  = (uint16_t*)scores;  // bf16 attn aliased onto score rows

  // 1. precision conversion / weight transposition
  cvt_bf16_kernel<<<(MSx * Ex) / 256, 256, 0, stream>>>(x, xbf, MSx * Ex);
  transpose_bf16_kernel<<<(Ex * Ex) / 256, 256, 0, stream>>>(W1, W1t, Ex, Ex);
  transpose_bf16_kernel<<<(Ex * Ex) / 256, 256, 0, stream>>>(W2, W2t, Ex, Ex);
  transpose_bf16_kernel<<<(Ex * Ex) / 256, 256, 0, stream>>>(W3, W3t, Ex, Ex);
  transpose_bf16_kernel<<<(Ex * FEx) / 256, 256, 0, stream>>>(Wf1, Wf1t, Ex, FEx);
  transpose_bf16_kernel<<<(FEx * Ex) / 256, 256, 0, stream>>>(Wf2, Wf2t, FEx, Ex);

  // 2. QKV projections: [4096x512] * [512x512]  (8192 tiles -> 1024 WGs)
  gemm_wmma_bf16<OUT_QK><<<dim3(1024, 1, 1), 256, 0, stream>>>(
      xbf, 0, Ex, W1t, 0, Ex, b1, Qb, 0, 0, MSx / 16, Ex / 16, Ex, 1.0f);
  gemm_wmma_bf16<OUT_QK><<<dim3(1024, 1, 1), 256, 0, stream>>>(
      xbf, 0, Ex, W2t, 0, Ex, b2, Kb, 0, 0, MSx / 16, Ex / 16, Ex, 1.0f);
  gemm_wmma_bf16<OUT_VT><<<dim3(1024, 1, 1), 256, 0, stream>>>(
      xbf, 0, Ex, W3t, 0, Ex, b3, Vt, 0, 0, MSx / 16, Ex / 16, Ex, 1.0f);

  // 3. scores = Q K^T / 8 : per (b,h) [2048x64] * [64x2048]
  gemm_wmma_bf16<OUT_F32><<<dim3(2048, 1, Bx * Hx), 256, 0, stream>>>(
      Qb, (long long)Sx * HDx, HDx,
      Kb, (long long)Sx * HDx, HDx,
      nullptr, scores, (long long)Sx * Sx, Sx,
      Sx / 16, Sx / 16, HDx, 0.125f);

  // 4. pot block: argmax -> gelu(mv@Wp+bp) -> substitute + softmax (bf16)
  rowmax_argmax_kernel<<<Bx * Hx * Sx, 256, 0, stream>>>(scores, mv, idx);
  pot_transform_kernel<<<Sx / 256, 256, 0, stream>>>(mv, Wp, bp, tv);
  softmax_sub_kernel<<<Bx * Hx * Sx, 256, 0, stream>>>(scores, attn, idx, tv);

  // 5. out = attn @ V : per (b,h) [2048x2048] * [2048x64], head-merged bf16
  gemm_wmma_bf16<OUT_MERGE><<<dim3(64, 1, Bx * Hx), 256, 0, stream>>>(
      attn, (long long)Sx * (2 * Sx), 2 * Sx,
      Vt, (long long)HDx * Sx, Sx,
      nullptr, attO, 0, 0, Sx / 16, HDx / 16, Sx, 1.0f);

  // 6. FFN: gelu(out@Wf1+bf1) @ Wf2 + bf2
  gemm_wmma_bf16<OUT_BF16_GELU><<<dim3(4096, 1, 1), 256, 0, stream>>>(
      attO, 0, Ex, Wf1t, 0, Ex, bf1, h1, 0, FEx, MSx / 16, FEx / 16, Ex, 1.0f);
  gemm_wmma_bf16<OUT_F32><<<dim3(1024, 1, 1), 256, 0, stream>>>(
      h1, 0, FEx, Wf2t, 0, FEx, bf2, ffn, 0, Ex, MSx / 16, Ex / 16, FEx, 1.0f);

  // 7. layernorm(ffn + x)
  add_ln_kernel<<<MSx, 256, 0, stream>>>(ffn, x, gamma, beta, (float*)d_out);
}
